// PrototypicalHead_45268955299907
// MI455X (gfx1250) — matrix operand order repaired and checked by reading, not compiled, not run on hardware
//
#include <hip/hip_runtime.h>

// ---------------------------------------------------------------------------
// PrototypicalHead for MI455X (gfx1250, wave32, WMMA).
//
// out[b,c] = sum_j max_l ( l2norm(spatial)[b,l,:] . l2norm(protos)[c,j,:] )
//            * softplus(raw_w[c,j]) + bias[c]
//
// GEMM M=6272 (b*l), N=10000 (c*j), K=768 -> 96 GFLOP, ~50MB traffic:
// compute-bound => v_wmma_f32_16x16x32_bf16 with split-bf16 (hi+lo) operands:
//   z.p ~= zh.ph + zh.pl + zl.ph   (3 WMMAs, ~fp32-accurate cosine sims)
//
// hi/lo planes are row-interleaved ([row][2][768]) so the lo fragment sits at
// a constant +1536B from the hi fragment and folds into the load offset field.
// 7 waves x 2 M-tile slots cover the 13 M tiles (1 clamped duplicate; max is
// idempotent) -> branch-free hot loop, EXEC all-ones through every WMMA.
// ---------------------------------------------------------------------------

typedef __attribute__((ext_vector_type(16))) __bf16 v16bf;
typedef __attribute__((ext_vector_type(8)))  __bf16 v8bf;
typedef __attribute__((ext_vector_type(8)))  float  v8f;

namespace {
constexpr int kB  = 32;
constexpr int kL  = 196;
constexpr int kLP = 208;          // 13 * 16 (padded l per batch)
constexpr int kD  = 768;
constexpr int kRS = 2 * kD;       // interleaved row stride (hi row, lo row)
constexpr int kC  = 1000;
constexpr int kJ  = 10;
constexpr int kN  = kC * kJ;      // 10000 flattened (c,j)
constexpr int kNT = kN / 16;      // 625 N tiles (exact)
constexpr int kMT = kLP / 16;     // 13 M tiles per batch
constexpr int kKS = kD / 32;      // 24 K steps
constexpr int kWaves = 7;         // 224 threads, 2 tile slots per wave
}

// --------------------------- out = bias broadcast ---------------------------
__global__ void init_out_kernel(const float* __restrict__ bias,
                                float* __restrict__ out) {
  int i = blockIdx.x * blockDim.x + threadIdx.x;
  if (i < kB * kC) out[i] = bias[i % kC];
}

// ----------------------- block-wide sum of squares (256t) -------------------
__device__ __forceinline__ float block_reduce_sum256(float ss) {
  __shared__ float red[8];
  #pragma unroll
  for (int o = 16; o > 0; o >>= 1) ss += __shfl_xor(ss, o, 32);
  const int t = threadIdx.x;
  if ((t & 31) == 0) red[t >> 5] = ss;
  __syncthreads();
  if (t < 32) {
    float x = (t < 8) ? red[t] : 0.0f;
    #pragma unroll
    for (int o = 4; o > 0; o >>= 1) x += __shfl_xor(x, o, 32);
    if (t == 0) red[0] = x;
  }
  __syncthreads();
  float tot = red[0];
  __syncthreads();
  return tot;
}

// ------ l2-normalize spatial rows -> interleaved bf16 hi/lo (padded) --------
__global__ void norm_spatial_kernel(const float* __restrict__ src,
                                    __bf16* __restrict__ dst) {
  const int row = blockIdx.x;            // [0, kB*kLP)
  const int b = row / kLP, l = row % kLP;
  const int t = threadIdx.x;             // 256 threads, kD = 3*256
  __bf16* hrow = dst + (size_t)row * kRS;
  __bf16* lrow = hrow + kD;
  if (l >= kL) {                         // deterministic zero padding rows
    for (int i = t; i < kD; i += 256) { hrow[i] = (__bf16)0.0f; lrow[i] = (__bf16)0.0f; }
    return;
  }
  const float* s = src + ((size_t)b * kL + l) * kD;
  float v0 = s[t], v1 = s[t + 256], v2 = s[t + 512];
  float tot = block_reduce_sum256(v0 * v0 + v1 * v1 + v2 * v2);
  float scale = 1.0f / fmaxf(sqrtf(tot), 1e-12f);
  float z; __bf16 h;
  z = v0 * scale; h = (__bf16)z; hrow[t]       = h; lrow[t]       = (__bf16)(z - (float)h);
  z = v1 * scale; h = (__bf16)z; hrow[t + 256] = h; lrow[t + 256] = (__bf16)(z - (float)h);
  z = v2 * scale; h = (__bf16)z; hrow[t + 512] = h; lrow[t + 512] = (__bf16)(z - (float)h);
}

// --------- l2-normalize prototype rows -> interleaved bf16 hi/lo ------------
__global__ void norm_proto_kernel(const float* __restrict__ src,
                                  __bf16* __restrict__ dst) {
  const int row = blockIdx.x;            // [0, kN)
  const int t = threadIdx.x;
  const float* s = src + (size_t)row * kD;
  __bf16* hrow = dst + (size_t)row * kRS;
  __bf16* lrow = hrow + kD;
  float v0 = s[t], v1 = s[t + 256], v2 = s[t + 512];
  float tot = block_reduce_sum256(v0 * v0 + v1 * v1 + v2 * v2);
  float scale = 1.0f / fmaxf(sqrtf(tot), 1e-12f);
  float z; __bf16 h;
  z = v0 * scale; h = (__bf16)z; hrow[t]       = h; lrow[t]       = (__bf16)(z - (float)h);
  z = v1 * scale; h = (__bf16)z; hrow[t + 256] = h; lrow[t + 256] = (__bf16)(z - (float)h);
  z = v2 * scale; h = (__bf16)z; hrow[t + 512] = h; lrow[t + 512] = (__bf16)(z - (float)h);
}

// pack two 8-wide halves into a 16-wide fragment without elementwise moves
union frag_pack {
  struct { v8bf lo, hi; } p;
  v16bf v;
};
__device__ __forceinline__ v16bf load_a_frag(const __bf16* p) {
  frag_pack u;
  u.p.lo = *(const v8bf*)(p);        // K = base .. base+7
  u.p.hi = *(const v8bf*)(p + 16);   // K = base+16 .. base+23
  return u.v;
}

// -------- WMMA GEMM + max-over-l + softplus-weighted scatter-add ------------
// block = 224 threads (7 waves x 2 M-tile slots); grid = (625 n-tiles, 32 b)
__global__ void __launch_bounds__(kWaves * 32)
simmax_wmma_kernel(const __bf16* __restrict__ z, const __bf16* __restrict__ p,
                   const float* __restrict__ rw, float* __restrict__ out) {
  const int nt   = blockIdx.x;
  const int b    = blockIdx.y;
  const int lane = threadIdx.x & 31;
  const int wave = threadIdx.x >> 5;     // 0..6
  const int r16  = lane & 15;
  const int hsel = lane >> 4;            // 0 or 1 (half-wave select)

  // B fragment (ISA B 32x16 bf16): lane = column n, K = hsel*8 + {0..7, 16..23}.
  // hi at +0, lo at +kD (folds into the load immediate offset).
  const __bf16* pb = p + (size_t)(nt * 16 + r16) * kRS + hsel * 8;

  // A fragment (ISA A 16x32 bf16): lane = row m, K = hsel*8 + {0..7, 16..23}.
  // Branch-free M-tile ownership: wave w owns tiles {w, w+7}; tile 13 clamps
  // to 12 (duplicate is harmless under the max reduction).
  int tclamp[2];
  v8f acc[2];
  const __bf16* ap[2];
  #pragma unroll
  for (int i = 0; i < 2; ++i) {
    v8f zero = {};
    acc[i] = zero;
    int t = wave + kWaves * i;
    tclamp[i] = (t < kMT) ? t : (kMT - 1);
    ap[i] = z + (size_t)(b * kLP + tclamp[i] * 16 + r16) * kRS + hsel * 8;
  }

  for (int kk = 0; kk < kKS; ++kk) {
    const int k0 = kk * 32;
    // speculative prefetch of next K chunk (silently dropped past the end)
    __builtin_prefetch(pb + k0 + 32, 0, 0);
    __builtin_prefetch(pb + kD + k0 + 32, 0, 0);

    v16bf bh = load_a_frag(pb + k0);
    v16bf bl = load_a_frag(pb + kD + k0);
    v16bf ah[2], al[2];
    #pragma unroll
    for (int i = 0; i < 2; ++i) {
      ah[i] = load_a_frag(ap[i] + k0);        // hi row
      al[i] = load_a_frag(ap[i] + kD + k0);   // lo row (+1536B, immediate)
    }
    // term-major issue order interleaves the two accumulator chains
    acc[0] = __builtin_amdgcn_wmma_f32_16x16x32_bf16(false, ah[0], false, bh, (short)0, acc[0], false, false);
    acc[1] = __builtin_amdgcn_wmma_f32_16x16x32_bf16(false, ah[1], false, bh, (short)0, acc[1], false, false);
    acc[0] = __builtin_amdgcn_wmma_f32_16x16x32_bf16(false, ah[0], false, bl, (short)0, acc[0], false, false);
    acc[1] = __builtin_amdgcn_wmma_f32_16x16x32_bf16(false, ah[1], false, bl, (short)0, acc[1], false, false);
    acc[0] = __builtin_amdgcn_wmma_f32_16x16x32_bf16(false, al[0], false, bh, (short)0, acc[0], false, false);
    acc[1] = __builtin_amdgcn_wmma_f32_16x16x32_bf16(false, al[1], false, bh, (short)0, acc[1], false, false);
  }

  // ---- max over valid l (C/D layout: m = hsel*8 + r, n = r16) ----
  float cmax = -3.402823466e+38f;
  #pragma unroll
  for (int i = 0; i < 2; ++i) {
    #pragma unroll
    for (int e = 0; e < 8; ++e) {
      int l = tclamp[i] * 16 + hsel * 8 + e;
      cmax = (l < kL) ? fmaxf(cmax, acc[i][e]) : cmax;
    }
  }
  // combine the two M-halves living in lanes (x) and (x^16); wave32 shuffle
  cmax = fmaxf(cmax, __shfl_xor(cmax, 16, 32));

  __shared__ float smax[kWaves][16];
  if (lane < 16) smax[wave][r16] = cmax;
  __syncthreads();

  if (threadIdx.x < 16) {
    float m = smax[0][threadIdx.x];
    #pragma unroll
    for (int w = 1; w < kWaves; ++w) m = fmaxf(m, smax[w][threadIdx.x]);
    int n = nt * 16 + threadIdx.x;       // flattened (c,j); always < kN (625*16)
    float x = rw[n];
    float w = fmaxf(x, 0.0f) + log1pf(__expf(-fabsf(x)));   // stable softplus
    atomicAdd(out + (size_t)b * kC + (n / kJ), m * w);      // global_atomic_add_f32
  }
}

// ---------------------------------------------------------------------------
extern "C" void kernel_launch(void* const* d_in, const int* in_sizes, int n_in,
                              void* d_out, int out_size, void* d_ws, size_t ws_size,
                              hipStream_t stream) {
  (void)in_sizes; (void)n_in; (void)out_size; (void)ws_size;

  const float* spatial = (const float*)d_in[0];   // [32,196,768]
  const float* protos  = (const float*)d_in[1];   // [1000,10,768]
  const float* rweight = (const float*)d_in[2];   // [1000,10]
  const float* bias    = (const float*)d_in[3];   // [1000]
  float* out = (float*)d_out;                     // [32,1000]

  // carve interleaved hi/lo bf16 planes out of the workspace (256B aligned)
  uintptr_t base = (uintptr_t)d_ws;
  auto take = [&](size_t bytes) -> void* {
    uintptr_t p = (base + 255) & ~(uintptr_t)255;
    base = p + bytes;
    return (void*)p;
  };
  __bf16* zbuf = (__bf16*)take((size_t)kB * kLP * kRS * sizeof(__bf16)); // ~20 MB
  __bf16* pbuf = (__bf16*)take((size_t)kN * kRS * sizeof(__bf16));       // ~31 MB

  init_out_kernel<<<(kB * kC + 255) / 256, 256, 0, stream>>>(bias, out);
  norm_spatial_kernel<<<kB * kLP, 256, 0, stream>>>(spatial, zbuf);
  norm_proto_kernel<<<kN, 256, 0, stream>>>(protos, pbuf);
  simmax_wmma_kernel<<<dim3(kNT, kB), kWaves * 32, 0, stream>>>(zbuf, pbuf, rweight, out);
}
